// RC_Explainer_Batch_71725953843766
// MI455X (gfx1250) — compile-verified, matrix operand order checked by reading, not run on hardware
//
#include <hip/hip_runtime.h>
#include <hip/hip_bf16.h>
#include <math.h>

// ---------------- problem constants (match reference) ----------------
#define N_NODES  50000
#define N_EDGES  400000
#define N_GRAPHS 500
#define FEAT     64
#define GOUT     32

typedef __attribute__((ext_vector_type(16))) _Float16 v16h;
typedef __attribute__((ext_vector_type(8)))  _Float16 v8h;
typedef __attribute__((ext_vector_type(8)))  float    v8f;

#define WMMA_F16(a,b,c) __builtin_amdgcn_wmma_f32_16x16x32_f16(false,(a),false,(b),(short)0,(c),false,false)

// ---------------- small helpers ----------------
__device__ inline float elu_f(float v) { return v > 0.f ? v : (__expf(v) - 1.f); }

__device__ inline void atomicMaxF(float* addr, float val) {
    unsigned int* ua = (unsigned int*)addr;
    unsigned int old = *ua;
    while (true) {
        float f = __uint_as_float(old);
        if (!(val > f)) break;
        unsigned int assumed = old;
        old = atomicCAS(ua, assumed, __float_as_uint(val));
        if (old == assumed) break;
    }
}

// ---------------- stage A: init / scatter / node reps ----------------
__global__ void init_agg_kernel(float* aggf, float* aggo, int n) {
    int t = blockIdx.x * blockDim.x + threadIdx.x;
    if (t < n) { aggf[t] = 0.f; aggo[t] = 0.f; }
}

__global__ void init_graph_kernel(float* m, float* denom, float* maxp, int* action) {
    int t = blockIdx.x * blockDim.x + threadIdx.x;
    if (t < N_GRAPHS) {
        m[t]     = -INFINITY;
        denom[t] = 0.f;
        maxp[t]  = -INFINITY;
        action[t] = N_EDGES;
    }
}

__global__ void scatter_kernel(const float* __restrict__ x,
                               const int* __restrict__ src,
                               const int* __restrict__ dst,
                               const unsigned char* __restrict__ state,
                               float* aggf, float* aggo) {
    int t = blockIdx.x * blockDim.x + threadIdx.x;
    if (t >= N_EDGES * FEAT) return;
    int e = t >> 6;
    int f = t & 63;
    int s = src[e], d = dst[e];
    float v = x[s * FEAT + f];
    atomicAdd(&aggf[d * FEAT + f], v);
    if (state[e]) atomicAdd(&aggo[d * FEAT + f], v);
}

__global__ void node_reps_kernel(const float* __restrict__ x,
                                 const float* __restrict__ aggf,
                                 const float* __restrict__ aggo,
                                 const float* __restrict__ Wg,
                                 const float* __restrict__ bg,
                                 _Float16* __restrict__ nreps) {
    int t = blockIdx.x * blockDim.x + threadIdx.x;
    if (t >= N_NODES * GOUT) return;
    int n = t >> 5;
    int o = t & 31;
    float s1 = 0.f, s2 = 0.f;
    #pragma unroll 8
    for (int k = 0; k < FEAT; ++k) {
        float w  = Wg[k * GOUT + o];
        float xa = x[n * FEAT + k];
        s1 += (xa + aggf[n * FEAT + k]) * w;
        s2 += (xa + aggo[n * FEAT + k]) * w;
    }
    float b = bg[o];
    nreps[t] = (_Float16)(elu_f(s1 + b) - elu_f(s2 + b));
}

// ---------------- stage B: pack weights into WMMA B-fragment order ----------------
// B tile is KxN = 32x16. Lane L supplies column n = L%16; halves follow ISA layout:
//   lanes 0-15 : K = {0..7, 16..23},  lanes 16-31 : K = {8..15, 24..31}
__global__ void pack_w_kernel(const float* __restrict__ W, _Float16* __restrict__ dst,
                              int KT, int NT, int Nreal) {
    int t = blockIdx.x * blockDim.x + threadIdx.x;
    int total = KT * NT * 512;
    if (t >= total) return;
    int h    = t & 15;
    int lane = (t >> 4) & 31;
    int tilei = t >> 9;
    int ktile = tilei % KT;
    int ntile = tilei / KT;
    int hi8 = (lane >> 4) * 8;
    int kl  = (h & 7) + hi8 + ((h >> 3) * 16);
    int k   = ktile * 32 + kl;
    int n   = ntile * 16 + (lane & 15);
    float v = (n < Nreal) ? W[k * Nreal + n] : 0.f;
    dst[((ntile * KT + ktile) * 32 + lane) * 16 + h] = (_Float16)v;
}

// ---------------- stage C: WMMA edge MLP ----------------
// One wave handles MROWS row-tiles (32 edges) so each B fragment feeds MROWS WMMAs,
// halving L2 weight traffic vs 1 tile/wave.
#define MROWS 2
#define EDGES_PER_WAVE (MROWS * 16)          // 32
#define NWTILES (N_EDGES / EDGES_PER_WAVE)   // 12500

__device__ inline v16h load_a_frag(const _Float16* rowp, int kOff) {
    v8h lo = *(const v8h*)(rowp + kOff);
    v8h hi = *(const v8h*)(rowp + kOff + 16);
    return __builtin_shufflevector(lo, hi, 0,1,2,3,4,5,6,7,8,9,10,11,12,13,14,15);
}

template<int KT, int NT, int DO_ELU>
__device__ inline void mlp_layer(const _Float16* inBuf,  // [32][KT*32] row-major (LDS)
                                 _Float16* outBuf,       // [32][NT*16] row-major (LDS)
                                 const _Float16* __restrict__ wfrag,
                                 const float* __restrict__ bias,
                                 int lane) {
    const int hi8 = (lane >> 4) * 8;
    const int col = lane & 15;
    v16h afr[MROWS][KT];
    #pragma unroll
    for (int mt = 0; mt < MROWS; ++mt) {
        const _Float16* rowp = inBuf + (mt * 16 + (lane & 15)) * (KT * 32);
        #pragma unroll
        for (int kt = 0; kt < KT; ++kt) afr[mt][kt] = load_a_frag(rowp, kt * 32 + hi8);
    }
    #pragma unroll
    for (int nt = 0; nt < NT; ++nt) {
        v8f acc[MROWS] = {};
        #pragma unroll
        for (int kt = 0; kt < KT; ++kt) {
            v16h bfr = *(const v16h*)(wfrag + ((nt * KT + kt) * 32 + lane) * 16);
            #pragma unroll
            for (int mt = 0; mt < MROWS; ++mt)
                acc[mt] = WMMA_F16(afr[mt][kt], bfr, acc[mt]);
        }
        int n = nt * 16 + col;
        float b = bias[n];
        #pragma unroll
        for (int mt = 0; mt < MROWS; ++mt) {
            #pragma unroll
            for (int r = 0; r < 8; ++r) {
                float v = acc[mt][r] + b;
                if (DO_ELU) v = elu_f(v);
                outBuf[(mt * 16 + r + hi8) * (NT * 16) + n] = (_Float16)v;
            }
        }
    }
}

__global__ void __launch_bounds__(32)
mlp_wmma_kernel(const _Float16* __restrict__ nreps,
                const int* __restrict__ src,
                const int* __restrict__ dst,
                const unsigned char* __restrict__ state,
                const int* __restrict__ batch,
                const int* __restrict__ y,
                const _Float16* __restrict__ wf1, const float* __restrict__ b1,
                const _Float16* __restrict__ wf2, const float* __restrict__ b2,
                const _Float16* __restrict__ wf3, const float* __restrict__ b3,
                const _Float16* __restrict__ wfp1, const float* __restrict__ pb1,
                const _Float16* __restrict__ wfp2, const float* __restrict__ pb2,
                float* __restrict__ lgbuf, int* __restrict__ gbuf) {
    __shared__ __align__(16) _Float16 sA[EDGES_PER_WAVE * 256];
    __shared__ __align__(16) _Float16 sB[EDGES_PER_WAVE * 256];
    __shared__ __align__(16) float    sL[EDGES_PER_WAVE * 16];

    const int lane = threadIdx.x & 31;
    const int e0   = blockIdx.x * EDGES_PER_WAVE;  // grid exactly covers NWTILES

    // stage 0: gather concat(node_reps[src], node_reps[dst]) -> sA [32][64]
    for (int i = lane; i < EDGES_PER_WAVE * 64; i += 32) {
        int row = i >> 6;
        int c   = i & 63;
        int e   = e0 + row;
        int node = (c < 32) ? src[e] : dst[e];
        sA[row * 64 + c] = nreps[node * GOUT + (c & 31)];
    }
    __syncthreads();

    mlp_layer<2, 16, 1>(sA, sB, wf1, b1, lane);  // 64 -> 256, ELU
    __syncthreads();
    mlp_layer<8, 8, 1>(sB, sA, wf2, b2, lane);   // 256 -> 128, ELU
    __syncthreads();
    mlp_layer<4, 4, 0>(sA, sB, wf3, b3, lane);   // 128 -> 64, linear
    __syncthreads();
    mlp_layer<2, 4, 1>(sB, sA, wfp1, pb1, lane); // 64 -> 64, ELU
    __syncthreads();

    // P2: 64 -> 16 (10 real labels, padded), f32 logits into sL
    {
        const int hi8 = (lane >> 4) * 8;
        const int col = lane & 15;
        float pb = (col < 10) ? pb2[col] : 0.f;
        v16h bb0 = *(const v16h*)(wfp2 + (0 * 32 + lane) * 16);
        v16h bb1 = *(const v16h*)(wfp2 + (1 * 32 + lane) * 16);
        #pragma unroll
        for (int mt = 0; mt < MROWS; ++mt) {
            const _Float16* rowp = sA + (mt * 16 + (lane & 15)) * 64;
            v16h a0 = load_a_frag(rowp, 0 + hi8);
            v16h a1 = load_a_frag(rowp, 32 + hi8);
            v8f acc = {};
            acc = WMMA_F16(a0, bb0, acc);
            acc = WMMA_F16(a1, bb1, acc);
            #pragma unroll
            for (int r = 0; r < 8; ++r)
                sL[(mt * 16 + r + hi8) * 16 + col] = acc[r] + pb;
        }
    }
    __syncthreads();

    // gather per-edge target-label logit, mask occupied edges (one lane per edge row)
    {
        int row = lane;
        int e   = e0 + row;
        int s   = src[e];
        int g   = batch[s];
        int lab = y[g];
        float lg = sL[row * 16 + lab];
        if (state[e]) lg = -INFINITY;
        lgbuf[e] = lg;
        gbuf[e]  = g;
    }
}

// ---------------- stage D: segment softmax + scatter max ----------------
__global__ void segmax_kernel(const float* __restrict__ lg, const int* __restrict__ g,
                              float* m) {
    int e = blockIdx.x * blockDim.x + threadIdx.x;
    if (e >= N_EDGES) return;
    atomicMaxF(&m[g[e]], lg[e]);
}

__global__ void expsum_kernel(const float* __restrict__ lg, const int* __restrict__ g,
                              const unsigned char* __restrict__ state,
                              const float* __restrict__ m,
                              float* __restrict__ ex, float* denom) {
    int e = blockIdx.x * blockDim.x + threadIdx.x;
    if (e >= N_EDGES) return;
    int gi = g[e];
    float v = state[e] ? 0.f : __expf(lg[e] - m[gi]);
    ex[e] = v;
    atomicAdd(&denom[gi], v);
}

__global__ void probs_kernel(const float* __restrict__ ex, const int* __restrict__ g,
                             const unsigned char* __restrict__ state,
                             const float* __restrict__ denom,
                             float* __restrict__ probs, float* maxp) {
    int e = blockIdx.x * blockDim.x + threadIdx.x;
    if (e >= N_EDGES) return;
    int gi = g[e];
    float p = ex[e] / denom[gi];
    probs[e] = p;
    float mp = state[e] ? -INFINITY : p;
    atomicMaxF(&maxp[gi], mp);
}

__global__ void argmin_kernel(const float* __restrict__ probs, const int* __restrict__ g,
                              const unsigned char* __restrict__ state,
                              const float* __restrict__ maxp, int* action) {
    int e = blockIdx.x * blockDim.x + threadIdx.x;
    if (e >= N_EDGES) return;
    int gi = g[e];
    float mp = state[e] ? -INFINITY : probs[e];
    if (mp == maxp[gi]) atomicMin(&action[gi], e);
}

__global__ void finalize_kernel(const float* __restrict__ maxp, const int* __restrict__ action,
                                float* __restrict__ out) {
    int t = blockIdx.x * blockDim.x + threadIdx.x;
    if (t >= N_GRAPHS) return;
    out[N_EDGES + t]            = maxp[t];
    out[N_EDGES + N_GRAPHS + t] = (float)action[t];
}

// ---------------- host-side launch ----------------
static inline size_t align_up(size_t v, size_t a) { return (v + a - 1) & ~(a - 1); }

extern "C" void kernel_launch(void* const* d_in, const int* in_sizes, int n_in,
                              void* d_out, int out_size, void* d_ws, size_t ws_size,
                              hipStream_t stream) {
    const float*         x     = (const float*)d_in[0];
    const int*           eidx  = (const int*)d_in[1];
    const unsigned char* state = (const unsigned char*)d_in[2];
    const int*           batch = (const int*)d_in[3];
    const int*           y     = (const int*)d_in[4];
    const float* Wg  = (const float*)d_in[5];
    const float* bg  = (const float*)d_in[6];
    const float* W1  = (const float*)d_in[7];
    const float* b1  = (const float*)d_in[8];
    const float* W2  = (const float*)d_in[9];
    const float* b2  = (const float*)d_in[10];
    const float* W3  = (const float*)d_in[11];
    const float* b3  = (const float*)d_in[12];
    const float* P1  = (const float*)d_in[13];
    const float* pb1 = (const float*)d_in[14];
    const float* P2  = (const float*)d_in[15];
    const float* pb2 = (const float*)d_in[16];

    const int* src = eidx;
    const int* dst = eidx + N_EDGES;

    // workspace carve-out
    char* ws = (char*)d_ws;
    size_t off = 0;
    float* aggf = (float*)(ws + off);        off = align_up(off + (size_t)N_NODES * FEAT * 4, 256);
    float* aggo = (float*)(ws + off);        off = align_up(off + (size_t)N_NODES * FEAT * 4, 256);
    _Float16* nreps = (_Float16*)(ws + off); off = align_up(off + (size_t)N_NODES * GOUT * 2, 256);
    // packed weight fragments (halves): L1 16384, L2 32768, L3 8192, P1 4096, P2 1024
    _Float16* wf1  = (_Float16*)(ws + off);  off = align_up(off + 16384 * 2, 256);
    _Float16* wf2  = (_Float16*)(ws + off);  off = align_up(off + 32768 * 2, 256);
    _Float16* wf3  = (_Float16*)(ws + off);  off = align_up(off + 8192 * 2, 256);
    _Float16* wfp1 = (_Float16*)(ws + off);  off = align_up(off + 4096 * 2, 256);
    _Float16* wfp2 = (_Float16*)(ws + off);  off = align_up(off + 1024 * 2, 256);
    float* lgbuf = (float*)(ws + off);       off = align_up(off + (size_t)N_EDGES * 4, 256);
    int*   gbuf  = (int*)(ws + off);         off = align_up(off + (size_t)N_EDGES * 4, 256);
    float* exbuf = (float*)(ws + off);       off = align_up(off + (size_t)N_EDGES * 4, 256);
    float* mbuf  = (float*)(ws + off);       off = align_up(off + N_GRAPHS * 4, 256);
    float* denom = (float*)(ws + off);       off = align_up(off + N_GRAPHS * 4, 256);
    float* maxp  = (float*)(ws + off);       off = align_up(off + N_GRAPHS * 4, 256);
    int*   action = (int*)(ws + off);        off = align_up(off + N_GRAPHS * 4, 256);

    float* out = (float*)d_out;

    // ---- stage A ----
    {
        int n = N_NODES * FEAT;
        init_agg_kernel<<<(n + 255) / 256, 256, 0, stream>>>(aggf, aggo, n);
        init_graph_kernel<<<(N_GRAPHS + 255) / 256, 256, 0, stream>>>(mbuf, denom, maxp, action);
        int m = N_EDGES * FEAT;
        scatter_kernel<<<(m + 255) / 256, 256, 0, stream>>>(x, src, dst, state, aggf, aggo);
        int r = N_NODES * GOUT;
        node_reps_kernel<<<(r + 255) / 256, 256, 0, stream>>>(x, aggf, aggo, Wg, bg, nreps);
    }

    // ---- stage B: pack weights ----
    {
        int t1 = 2 * 16 * 512;  pack_w_kernel<<<(t1 + 255) / 256, 256, 0, stream>>>(W1, wf1, 2, 16, 256);
        int t2 = 8 * 8  * 512;  pack_w_kernel<<<(t2 + 255) / 256, 256, 0, stream>>>(W2, wf2, 8, 8, 128);
        int t3 = 4 * 4  * 512;  pack_w_kernel<<<(t3 + 255) / 256, 256, 0, stream>>>(W3, wf3, 4, 4, 64);
        int t4 = 2 * 4  * 512;  pack_w_kernel<<<(t4 + 255) / 256, 256, 0, stream>>>(P1, wfp1, 2, 4, 64);
        int t5 = 2 * 1  * 512;  pack_w_kernel<<<(t5 + 255) / 256, 256, 0, stream>>>(P2, wfp2, 2, 1, 10);
    }

    // ---- stage C: WMMA MLP over all edges (one wave per 32 edges) ----
    mlp_wmma_kernel<<<NWTILES, 32, 0, stream>>>(
        nreps, src, dst, state, batch, y,
        wf1, b1, wf2, b2, wf3, b3, wfp1, pb1, wfp2, pb2,
        lgbuf, gbuf);

    // ---- stage D: segment softmax + scatter max ----
    {
        int gsz = (N_EDGES + 255) / 256;
        segmax_kernel<<<gsz, 256, 0, stream>>>(lgbuf, gbuf, mbuf);
        expsum_kernel<<<gsz, 256, 0, stream>>>(lgbuf, gbuf, state, mbuf, exbuf, denom);
        probs_kernel<<<gsz, 256, 0, stream>>>(exbuf, gbuf, state, denom, out, maxp);
        argmin_kernel<<<gsz, 256, 0, stream>>>(out, gbuf, state, maxp, action);
        finalize_kernel<<<(N_GRAPHS + 255) / 256, 256, 0, stream>>>(maxp, action, out);
    }
}
